// GCN_7602092113943
// MI455X (gfx1250) — compile-verified, matrix operand order checked by reading, not run on hardware
//
#include <hip/hip_runtime.h>
#include <hip/hip_bf16.h>
#include <math.h>

// GCN forward for MI455X (gfx1250, wave32).
//   h1 = relu( Agg( x @ W1 ) + b1 );  out = log_softmax( Agg( h1 @ W2 ) + b2 )
// GEMMs use v_wmma_f32_16x16x32_bf16 (bf16 in, f32 accumulate).
// GEMM1: one wave per 16-row m-tile computes ALL 4 n-tiles (A loaded once per
// k-step, 4 WMMAs) so x (205 MB > 192 MB L2) is read from HBM exactly once.
// Aggregation = per-edge gather + global f32 atomic scatter-add (memory bound,
// ~2.1 GB total traffic -> ~90us at 23.3 TB/s; GEMM FLOPs are negligible).

typedef __attribute__((ext_vector_type(16))) __bf16 v16bf;
typedef __attribute__((ext_vector_type(8)))  __bf16 v8bf;
typedef __attribute__((ext_vector_type(8)))  float  v8f;
typedef __attribute__((ext_vector_type(4)))  float  v4f;   // native vector for nontemporal builtins

#define IN_F   512
#define HID_F  64
#define OUT_F  16

static __device__ __forceinline__ __bf16 f2bf(float f) { return (__bf16)f; }

// K index inside a 32-wide K-tile for packed 16-bit A/B fragments (wave32):
// lanes 0-15 cover K {0..7, 16..23}, lanes 16-31 cover K {8..15, 24..31}.
static __device__ __forceinline__ int kmap(int i, int hi) {
    return (i < 8) ? (hi * 8 + i) : (16 + hi * 8 + (i - 8));
}

// ---------------------------------------------------------------- utilities
__global__ void gcn_zero(float* __restrict__ p, long long n) {
    long long t = (long long)blockIdx.x * blockDim.x + threadIdx.x;
    if (t < n) p[t] = 0.0f;
}

// Pack fp32 row-major W[K x Nw] into bf16 WMMA B-fragment order:
// dst[(((ntile*(K/32) + ks)*32 + lane)*16 + i)] = W[ks*32 + kmap(i,hi)][ntile*16 + lane%16]
__global__ void gcn_pack_w(const float* __restrict__ w, __bf16* __restrict__ wp,
                           int K, int Nw) {
    int t = blockIdx.x * blockDim.x + threadIdx.x;
    int total = (Nw / 16) * (K / 32) * 32 * 16;
    if (t >= total) return;
    int i     = t & 15;
    int lane  = (t >> 4) & 31;
    int rest  = t >> 9;
    int ks    = rest % (K / 32);
    int ntile = rest / (K / 32);
    int hi    = lane >> 4;
    int k     = ks * 32 + kmap(i, hi);
    int n     = ntile * 16 + (lane & 15);
    wp[t] = f2bf(w[(size_t)k * Nw + n]);
}

__global__ void gcn_degree(const long long* __restrict__ dst, float* __restrict__ deg,
                           long long n_edges) {
    long long t = (long long)blockIdx.x * blockDim.x + threadIdx.x;
    if (t < n_edges) atomicAdd(&deg[dst[t]], 1.0f);
}

__global__ void gcn_dinv(const float* __restrict__ deg, float* __restrict__ dinv, int n) {
    int i = blockIdx.x * blockDim.x + threadIdx.x;
    if (i < n) dinv[i] = rsqrtf(deg[i] + 1.0f);   // +1 self-loop; always > 0
}

// ---------------------------------------------------------------- GEMM 1
// p1[N x 64] = x[N x 512] @ W1[512 x 64].
// 256 threads = 8 waves; each wave owns one 16-row m-tile and produces all
// four 16-col n-tiles: A fragment loaded once per k-step, reused by 4 WMMAs.
__global__ __launch_bounds__(256) void gcn_gemm1(
        const float* __restrict__ x, const __bf16* __restrict__ w1p,
        float* __restrict__ p1, int n_nodes) {
    const int lane = threadIdx.x & 31;
    const int wave = threadIdx.x >> 5;
    const int m0   = (blockIdx.x * 8 + wave) * 16;
    if (m0 >= n_nodes) return;                  // wave-uniform exit (EXEC stays full)
    const int hi   = lane >> 4;
    const int lmod = lane & 15;
    int m = m0 + lmod;
    if (m >= n_nodes) m = n_nodes - 1;          // clamp; stores masked below
    const size_t rowbase = (size_t)m * IN_F;

    v8f acc[4] = {};
    #pragma unroll 2
    for (int ks = 0; ks < IN_F / 32; ++ks) {
        const v4f* xr = (const v4f*)(x + rowbase + ks * 32 + hi * 8);
        // x is streamed exactly once: non-temporal so it does not evict the
        // L2 lines (p1/h/p2) that the gather phases re-read randomly.
        v4f r0 = __builtin_nontemporal_load(xr);
        v4f r1 = __builtin_nontemporal_load(xr + 1);
        v4f r2 = __builtin_nontemporal_load(xr + 4);
        v4f r3 = __builtin_nontemporal_load(xr + 5);
        v16bf a;
        a[0]  = f2bf(r0.x); a[1]  = f2bf(r0.y); a[2]  = f2bf(r0.z); a[3]  = f2bf(r0.w);
        a[4]  = f2bf(r1.x); a[5]  = f2bf(r1.y); a[6]  = f2bf(r1.z); a[7]  = f2bf(r1.w);
        a[8]  = f2bf(r2.x); a[9]  = f2bf(r2.y); a[10] = f2bf(r2.z); a[11] = f2bf(r2.w);
        a[12] = f2bf(r3.x); a[13] = f2bf(r3.y); a[14] = f2bf(r3.z); a[15] = f2bf(r3.w);
        #pragma unroll
        for (int nt = 0; nt < 4; ++nt) {
            v16bf b = *(const v16bf*)(w1p +
                       (((size_t)(nt * (IN_F / 32) + ks)) * 32 + lane) * 16);
            acc[nt] = __builtin_amdgcn_wmma_f32_16x16x32_bf16(
                          false, a, false, b, (short)0, acc[nt], false, false);
        }
    }
    #pragma unroll
    for (int nt = 0; nt < 4; ++nt) {
        const int col = nt * 16 + lmod;
        #pragma unroll
        for (int v = 0; v < 8; ++v) {
            int row = m0 + hi * 8 + v;          // C/D layout: VGPR v -> M=v / v+8
            if (row < n_nodes) p1[(size_t)row * HID_F + col] = acc[nt][v];
        }
    }
}

// ---------------------------------------------------------------- scatter F=64
// One wave per edge; each lane handles 2 features (float2 gather, 2 atomics).
__global__ void gcn_agg64(const long long* __restrict__ src,
                          const long long* __restrict__ dst,
                          const float* __restrict__ dinv,
                          const float* __restrict__ p1,
                          float* __restrict__ agg, long long n_edges) {
    long long t = (long long)blockIdx.x * blockDim.x + threadIdx.x;
    long long e = t >> 5;
    if (e >= n_edges) return;
    int f2 = ((int)t & 31) * 2;
    long long s = src[e], d = dst[e];
    float w = dinv[s] * dinv[d];
    float2 v = *(const float2*)(p1 + (size_t)s * HID_F + f2);
    float* ar = agg + (size_t)d * HID_F + f2;
    atomicAdd(ar,     v.x * w);
    atomicAdd(ar + 1, v.y * w);
}

// self-loop + bias + ReLU, emit bf16 h for GEMM2
__global__ void gcn_relu_bias(const float* __restrict__ agg1,
                              const float* __restrict__ p1,
                              const float* __restrict__ dinv,
                              const float* __restrict__ b1,
                              __bf16* __restrict__ h, int n_nodes) {
    long long t = (long long)blockIdx.x * blockDim.x + threadIdx.x;
    if (t >= (long long)n_nodes * HID_F) return;
    int node = (int)(t >> 6);
    int f    = (int)(t & 63);
    float di = dinv[node];
    float v  = agg1[t] + p1[t] * di * di + b1[f];
    h[t] = f2bf(fmaxf(v, 0.0f));
}

// ---------------------------------------------------------------- GEMM 2
// p2[N x 16] = h[N x 64] @ W2[64 x 16]. 8 waves/block, one 16-row tile each.
__global__ __launch_bounds__(256) void gcn_gemm2(
        const __bf16* __restrict__ h, const __bf16* __restrict__ w2p,
        float* __restrict__ p2, int n_nodes) {
    const int lane = threadIdx.x & 31;
    const int wave = threadIdx.x >> 5;
    const int m0   = (blockIdx.x * 8 + wave) * 16;
    if (m0 >= n_nodes) return;                  // wave-uniform exit (EXEC stays full)
    const int hi   = lane >> 4;
    const int lmod = lane & 15;
    int m = m0 + lmod;
    if (m >= n_nodes) m = n_nodes - 1;
    const size_t rb = (size_t)m * HID_F;

    v8f acc = {};
    #pragma unroll
    for (int ks = 0; ks < HID_F / 32; ++ks) {
        const __bf16* hr = h + rb + ks * 32 + hi * 8;
        v8bf lo = *(const v8bf*)(hr);
        v8bf up = *(const v8bf*)(hr + 16);
        v16bf a = __builtin_shufflevector(lo, up,
                      0, 1, 2, 3, 4, 5, 6, 7, 8, 9, 10, 11, 12, 13, 14, 15);
        v16bf b = *(const v16bf*)(w2p + ((size_t)ks * 32 + lane) * 16);
        acc = __builtin_amdgcn_wmma_f32_16x16x32_bf16(
                  false, a, false, b, (short)0, acc, false, false);
    }
    #pragma unroll
    for (int v = 0; v < 8; ++v) {
        int row = m0 + hi * 8 + v;
        if (row < n_nodes) p2[(size_t)row * OUT_F + lmod] = acc[v];
    }
}

// ---------------------------------------------------------------- scatter F=16
__global__ void gcn_agg16(const long long* __restrict__ src,
                          const long long* __restrict__ dst,
                          const float* __restrict__ dinv,
                          const float* __restrict__ p2,
                          float* __restrict__ agg, long long n_edges) {
    long long t = (long long)blockIdx.x * blockDim.x + threadIdx.x;
    long long e = t >> 4;
    if (e >= n_edges) return;
    int f = (int)t & 15;
    long long s = src[e], d = dst[e];
    float w = dinv[s] * dinv[d];
    atomicAdd(agg + (size_t)d * OUT_F + f, p2[(size_t)s * OUT_F + f] * w);
}

// self-loop + bias + log_softmax over 16 features, one thread per node
__global__ void gcn_finalize(const float* __restrict__ agg2,
                             const float* __restrict__ p2,
                             const float* __restrict__ dinv,
                             const float* __restrict__ b2,
                             float* __restrict__ out, int n_nodes) {
    int i = blockIdx.x * blockDim.x + threadIdx.x;
    if (i >= n_nodes) return;
    float di = dinv[i], dd = di * di;
    const float* a = agg2 + (size_t)i * OUT_F;
    const float* p = p2   + (size_t)i * OUT_F;
    float vals[OUT_F];
    float mx = -3.402823466e38f;
    #pragma unroll
    for (int f = 0; f < OUT_F; ++f) {
        float v = a[f] + p[f] * dd + b2[f];
        vals[f] = v;
        mx = fmaxf(mx, v);
    }
    float s = 0.0f;
    #pragma unroll
    for (int f = 0; f < OUT_F; ++f) s += expf(vals[f] - mx);
    float lse = mx + logf(s);
    float* o = out + (size_t)i * OUT_F;
    #pragma unroll
    for (int f = 0; f < OUT_F; ++f) o[f] = vals[f] - lse;
}

// ---------------------------------------------------------------- launch
extern "C" void kernel_launch(void* const* d_in, const int* in_sizes, int n_in,
                              void* d_out, int out_size, void* d_ws, size_t ws_size,
                              hipStream_t stream) {
    (void)n_in; (void)out_size; (void)ws_size;
    const float*     x  = (const float*)d_in[0];
    const long long* ei = (const long long*)d_in[1];   // int64 edge_index [2, E]
    const float*     W1 = (const float*)d_in[2];
    const float*     b1 = (const float*)d_in[3];
    const float*     W2 = (const float*)d_in[4];
    const float*     b2 = (const float*)d_in[5];
    float* out = (float*)d_out;

    const int       N = in_sizes[0] / IN_F;            // 100000
    const long long E = (long long)(in_sizes[1] / 2);  // 3200000
    const long long* src = ei;
    const long long* dst = ei + E;

    // workspace carve-up (256 B aligned); deg/agg1/agg2 first -> one zero pass
    char* base = (char*)d_ws;
    size_t off = 0;
    auto carve = [&](size_t bytes) -> void* {
        off = (off + 255) & ~(size_t)255;
        void* p = base + off;
        off += bytes;
        return p;
    };
    float*  deg  = (float*)carve((size_t)N * 4);
    float*  agg1 = (float*)carve((size_t)N * HID_F * 4);
    float*  agg2 = (float*)carve((size_t)N * OUT_F * 4);
    size_t zero_bytes = (off + 255) & ~(size_t)255;
    float*  dinv = (float*)carve((size_t)N * 4);
    float*  p1   = (float*)carve((size_t)N * HID_F * 4);
    float*  p2   = (float*)carve((size_t)N * OUT_F * 4);
    __bf16* w1p  = (__bf16*)carve((size_t)IN_F * HID_F * 2);
    __bf16* w2p  = (__bf16*)carve((size_t)HID_F * OUT_F * 2);
    __bf16* h    = (__bf16*)carve((size_t)N * HID_F * 2);
    (void)deg;

    const int T = 256;
    auto nb = [&](long long work) -> unsigned {
        return (unsigned)((work + T - 1) / T);
    };

    long long zn = (long long)(zero_bytes / 4);
    gcn_zero<<<nb(zn), T, 0, stream>>>((float*)base, zn);

    gcn_pack_w<<<nb((HID_F / 16) * (IN_F / 32) * 512), T, 0, stream>>>(W1, w1p, IN_F, HID_F);
    gcn_pack_w<<<nb((OUT_F / 16) * (HID_F / 32) * 512), T, 0, stream>>>(W2, w2p, HID_F, OUT_F);

    gcn_degree<<<nb(E), T, 0, stream>>>(dst, (float*)base, E);
    gcn_dinv<<<nb(N), T, 0, stream>>>((const float*)base, dinv, N);

    unsigned g1 = (unsigned)(((N + 15) / 16 + 7) / 8);   // 8 m-tiles per block
    gcn_gemm1<<<g1, 256, 0, stream>>>(x, w1p, p1, N);

    gcn_agg64<<<nb(E * 32), T, 0, stream>>>(src, dst, dinv, p1, agg1, E);
    gcn_relu_bias<<<nb((long long)N * HID_F), T, 0, stream>>>(agg1, p1, dinv, b1, h, N);

    unsigned g2 = (unsigned)(((N + 15) / 16 + 7) / 8);   // 8 m-tiles per block
    gcn_gemm2<<<g2, 256, 0, stream>>>(h, w2p, p2, N);

    gcn_agg16<<<nb(E * 16), T, 0, stream>>>(src, dst, dinv, p2, agg2, E);
    gcn_finalize<<<nb(N), T, 0, stream>>>(agg2, p2, dinv, b2, out, N);
}
